// LSTMLayer_59665685676618
// MI455X (gfx1250) — matrix operand order repaired and checked by reading, not compiled, not run on hardware
//
#include <hip/hip_runtime.h>

typedef __attribute__((ext_vector_type(16))) __bf16 v16bf;
typedef __attribute__((ext_vector_type(8)))  float  v8f;
typedef __attribute__((ext_vector_type(4)))  int    i32x4;

#define T_STEPS 512
#define BATCH   64
#define IN_DIM  512
#define HID     512
#define G4      2048          // 4*HID
#define MROWS   (T_STEPS * BATCH)

#if defined(__AMDGCN__) && __has_builtin(__builtin_amdgcn_global_load_async_to_lds_b128) && \
    __has_builtin(__builtin_amdgcn_s_wait_asynccnt)
#define HAVE_ASYNC_LDS 1
#else
#define HAVE_ASYNC_LDS 0
#endif

#if HAVE_ASYNC_LDS
// Prototype (from hipcc diagnostic): (int4 AS1*, int4 AS3*, imm offset, imm cpol)
#define ASYNC_B128(gptr, lptr)                                                   \
  __builtin_amdgcn_global_load_async_to_lds_b128(                                \
      (__attribute__((address_space(1))) i32x4*)(gptr),                          \
      (__attribute__((address_space(3))) i32x4*)(lptr), 0, 0)
#define ASYNC_WAIT() __builtin_amdgcn_s_wait_asynccnt(0)
#else
#define ASYNC_B128(gptr, lptr) (*(uint4*)(lptr) = *(const uint4*)(gptr))
#define ASYNC_WAIT() ((void)0)
#endif

__device__ __forceinline__ unsigned short f2bf(float f) {
  unsigned int u = __float_as_uint(f);
  u += 0x7FFFu + ((u >> 16) & 1u);   // round-to-nearest-even
  return (unsigned short)(u >> 16);
}

__device__ __forceinline__ float sigmoidf_(float x) {
  return 1.0f / (1.0f + __expf(-x));
}

// A-matrix (16x32 bf16) fragment, K order matching B:
// lanes 0-15 (M=lane): K=0..15 ; lanes 16-31 (M=lane-16): K=16..31
__device__ __forceinline__ v16bf load_a_frag(const unsigned short* base, int stride,
                                             int row, int lane) {
  union { v16bf v; unsigned short u[16]; } x;
  const unsigned short* r = base + row * stride;
  const int ko = (lane < 16) ? 0 : 16;
#pragma unroll
  for (int e = 0; e < 16; ++e) {
    const int k = e + ko;
    x.u[e] = r[k];
  }
  return x.v;
}

// B-matrix (32x16 bf16), staged in LDS as [n][k]:
// lanes 0-15: N=lane, K=0..15 ; lanes 16-31: N=lane-16, K=16..31
__device__ __forceinline__ v16bf load_b_frag(const unsigned short* base, int stride,
                                             int row, int lane) {
  union { v16bf v; unsigned short u[16]; } x;
  const unsigned short* r = base + row * stride;
  const int ko = (lane < 16) ? 0 : 16;
#pragma unroll
  for (int e = 0; e < 16; ++e) x.u[e] = r[e + ko];
  return x.v;
}

// ---------------------------------------------------------------------------
// Prep: W_ih/W_hh fp32 [4H,K] -> bf16 in K-chunk-blocked B layout
//   WB[kc][n][kk] , kc = k/32, kk = k%32  (each column's K-chunk = 64B run)
// bias = b_ih + b_hh; seed bf16 h0 and fp32 c0.
// ---------------------------------------------------------------------------
__global__ __launch_bounds__(256) void lstm_prep(
    const float* __restrict__ Wih, const float* __restrict__ Whh,
    const float* __restrict__ bih, const float* __restrict__ bhh,
    const float* __restrict__ h0,  const float* __restrict__ c0,
    unsigned short* __restrict__ WihB, unsigned short* __restrict__ WhhB,
    float* __restrict__ bias, unsigned short* __restrict__ hb0,
    float* __restrict__ cbuf) {
  const int idx = blockIdx.x * blockDim.x + threadIdx.x;
  if (idx < G4 * IN_DIM) {
    const int kc  = idx / (G4 * 32);
    const int rem = idx % (G4 * 32);
    const int n   = rem / 32;
    const int kk  = rem % 32;
    const int k   = kc * 32 + kk;
    WihB[idx] = f2bf(Wih[n * IN_DIM + k]);
    WhhB[idx] = f2bf(Whh[n * HID + k]);
  }
  if (idx < G4) bias[idx] = bih[idx] + bhh[idx];
  if (idx < BATCH * HID) {
    hb0[idx] = f2bf(h0[idx]);
    cbuf[idx] = c0[idx];
  }
}

// ---------------------------------------------------------------------------
// x_gates GEMM: C[32768,2048] = input[32768,512] * W_ih^T + bias
// WG tile 128x128, 8 waves, each wave 16(M) x 128(N) = 8 wmma accumulators.
// B tile staged via async global->LDS b128 (blocked weight layout).
// ---------------------------------------------------------------------------
__global__ __launch_bounds__(256) void xgates_gemm(
    const float* __restrict__ A, const unsigned short* __restrict__ Bw,
    const float* __restrict__ bias, float* __restrict__ C) {
  __shared__ unsigned short sA[128 * 40];   // [m][k] bf16, padded stride
  __shared__ unsigned short sB[128 * 40];   // [n][k] bf16
  const int tid = threadIdx.x;
  const int lane = tid & 31, wave = tid >> 5;
  const int bm = blockIdx.x * 128;
  const int bn = blockIdx.y * 128;
  const int mw = wave * 16;

  v8f acc[8] = {};

  for (int k0 = 0; k0 < IN_DIM; k0 += 32) {
    __syncthreads();
    // B tile: blocked layout -> [n][k] LDS; 2 x b128 per thread, coalesced.
    {
      const int n    = tid >> 1;            // 0..127
      const int part = (tid & 1) * 16;      // bf16 element offset within column
      const unsigned short* g =
          Bw + ((size_t)(k0 >> 5) * G4 + bn + n) * 32 + part;
      ASYNC_B128(g,     &sB[n * 40 + part]);
      ASYNC_B128(g + 8, &sB[n * 40 + part + 8]);
    }
    // A tile 128x32: fp32 -> bf16 (needs conversion; manual path)
#pragma unroll
    for (int p = 0; p < 4; ++p) {
      const int row = p * 32 + (tid >> 3);
      const int col = (tid & 7) * 4;
      const float4 f = *(const float4*)(A + (size_t)(bm + row) * IN_DIM + k0 + col);
      unsigned short* d = &sA[row * 40 + col];
      d[0] = f2bf(f.x); d[1] = f2bf(f.y); d[2] = f2bf(f.z); d[3] = f2bf(f.w);
    }
    ASYNC_WAIT();
    __syncthreads();
    const v16bf a = load_a_frag(sA, 40, mw + (lane & 15), lane);
#pragma unroll
    for (int j = 0; j < 8; ++j) {
      const v16bf b = load_b_frag(sB, 40, j * 16 + (lane & 15), lane);
      acc[j] = __builtin_amdgcn_wmma_f32_16x16x32_bf16(
          false, a, false, b, (short)0, acc[j], false, false);
    }
  }
  // D layout: V v -> M = v (lanes 0-15) / v+8 (lanes 16-31); N = lane%16
  const int mbase = bm + mw + ((lane >> 4) * 8);
#pragma unroll
  for (int j = 0; j < 8; ++j) {
    const int ncol = bn + j * 16 + (lane & 15);
    const float bv = bias[ncol];
#pragma unroll
    for (int v = 0; v < 8; ++v)
      C[(size_t)(mbase + v) * G4 + ncol] = acc[j][v] + bv;
  }
}

// ---------------------------------------------------------------------------
// One recurrent step. Each WG owns 32 h-columns across all 4 gates:
// gates[64, {g*512 + nb..nb+31}] = h_prev @ Whh_cols, then fused LSTM cell.
// grid = 16 WGs, 8 waves; wave -> (mt in {0,16,32,48}) x (nt in {0,16}),
// 4 accumulators (one per gate). Both tiles staged with async b128 copies.
// ---------------------------------------------------------------------------
__global__ __launch_bounds__(256) void lstm_step(
    const unsigned short* __restrict__ hprev,   // [64,512] bf16
    const unsigned short* __restrict__ Whh,     // blocked [K/32][2048][32] bf16
    const float* __restrict__ xg,               // [64,2048] fp32 (this t)
    float* __restrict__ cbuf,                   // [64,512] fp32 state
    float* __restrict__ hout,                   // [64,512] fp32 -> d_out[t]
    unsigned short* __restrict__ hnext) {       // [64,512] bf16 ping-pong
  __shared__ unsigned short sA[64 * 40];        // h chunk  [b][k]
  __shared__ unsigned short sB[128 * 40];       // [4*32 gate-cols][k]
  const int tid  = threadIdx.x;
  const int lane = tid & 31, wave = tid >> 5;
  const int nb = blockIdx.x * 32;               // h-column base
  const int mt = (wave & 3) * 16;
  const int nt = (wave >> 2) * 16;

  v8f acc[4] = {};

  for (int k0 = 0; k0 < HID; k0 += 32) {
    __syncthreads();
    {   // h chunk 64x32 bf16: 1 x b128 per thread
      const int r  = tid >> 2;
      const int kk = (tid & 3) * 8;
      ASYNC_B128(hprev + (size_t)r * HID + k0 + kk, &sA[r * 40 + kk]);
    }
    {   // Whh chunk: 128 gate-cols x 32 k (blocked layout): 2 x b128 per thread
      const int nl   = tid >> 1;              // 0..127 local gate-column
      const int part = (tid & 1) * 16;
      const int gcol = (nl >> 5) * HID + nb + (nl & 31);
      const unsigned short* g =
          Whh + ((size_t)(k0 >> 5) * G4 + gcol) * 32 + part;
      ASYNC_B128(g,     &sB[nl * 40 + part]);
      ASYNC_B128(g + 8, &sB[nl * 40 + part + 8]);
      if (k0 + 32 < HID)   // prefetch next K-chunk (global_prefetch_b8)
        __builtin_prefetch(Whh + ((size_t)((k0 >> 5) + 1) * G4 + gcol) * 32, 0, 1);
    }
    ASYNC_WAIT();
    __syncthreads();
    const v16bf a = load_a_frag(sA, 40, mt + (lane & 15), lane);
#pragma unroll
    for (int g = 0; g < 4; ++g) {
      const v16bf b = load_b_frag(sB, 40, g * 32 + nt + (lane & 15), lane);
      acc[g] = __builtin_amdgcn_wmma_f32_16x16x32_bf16(
          false, a, false, b, (short)0, acc[g], false, false);
    }
  }

  // Fused LSTM cell on the accumulators (all 4 gates co-resident per lane).
  const int cl = nt + (lane & 15);
  const int nh = nb + cl;                        // h column (0..511)
  const int mbase = mt + ((lane >> 4) * 8);
#pragma unroll
  for (int v = 0; v < 8; ++v) {
    const int row = mbase + v;
    const size_t xrow = (size_t)row * G4;
    const float gi = acc[0][v] + xg[xrow + 0 * HID + nh];
    const float gf = acc[1][v] + xg[xrow + 1 * HID + nh];
    const float gg = acc[2][v] + xg[xrow + 2 * HID + nh];
    const float go = acc[3][v] + xg[xrow + 3 * HID + nh];
    const float i  = sigmoidf_(gi);
    const float f  = sigmoidf_(gf);
    const float gv = tanhf(gg);
    const float o  = sigmoidf_(go);
    const size_t sidx = (size_t)row * HID + nh;
    const float cn = f * cbuf[sidx] + i * gv;
    cbuf[sidx] = cn;
    const float hn = o * tanhf(cn);
    hout[sidx]  = hn;
    hnext[sidx] = f2bf(hn);
  }
}

// ---------------------------------------------------------------------------
extern "C" void kernel_launch(void* const* d_in, const int* in_sizes, int n_in,
                              void* d_out, int out_size, void* d_ws, size_t ws_size,
                              hipStream_t stream) {
  (void)in_sizes; (void)n_in; (void)out_size; (void)ws_size;
  const float* input = (const float*)d_in[0];
  const float* h0    = (const float*)d_in[1];
  const float* c0    = (const float*)d_in[2];
  const float* Wih   = (const float*)d_in[3];
  const float* Whh   = (const float*)d_in[4];
  const float* bih   = (const float*)d_in[5];
  const float* bhh   = (const float*)d_in[6];
  float* out = (float*)d_out;

  char* ws = (char*)d_ws;
  float* xg = (float*)ws;                                      // 256 MiB
  size_t off = (size_t)MROWS * G4 * 4;
  unsigned short* WihB = (unsigned short*)(ws + off); off += (size_t)IN_DIM * G4 * 2;
  unsigned short* WhhB = (unsigned short*)(ws + off); off += (size_t)HID * G4 * 2;
  float* bias = (float*)(ws + off);                   off += (size_t)G4 * 4;
  unsigned short* hb0 = (unsigned short*)(ws + off);  off += (size_t)BATCH * HID * 2;
  unsigned short* hb1 = (unsigned short*)(ws + off);  off += (size_t)BATCH * HID * 2;
  float* cbuf = (float*)(ws + off);                   off += (size_t)BATCH * HID * 4;

  lstm_prep<<<(G4 * IN_DIM + 255) / 256, 256, 0, stream>>>(
      Wih, Whh, bih, bhh, h0, c0, WihB, WhhB, bias, hb0, cbuf);

  xgates_gemm<<<dim3(MROWS / 128, G4 / 128), 256, 0, stream>>>(input, WihB, bias, xg);

  for (int t = 0; t < T_STEPS; ++t) {
    const unsigned short* hin = (t & 1) ? hb1 : hb0;
    unsigned short*       hnx = (t & 1) ? hb0 : hb1;
    lstm_step<<<HID / 32, 256, 0, stream>>>(
        hin, WhhB, xg + (size_t)t * BATCH * G4, cbuf,
        out + (size_t)t * BATCH * HID, hnx);
  }

  // Tail: hT = outputs[T-1], cT = final cell state.
  float* hT = out + (size_t)T_STEPS * BATCH * HID;
  float* cT = hT + (size_t)BATCH * HID;
  (void)hipMemcpyAsync(hT, out + (size_t)(T_STEPS - 1) * BATCH * HID,
                       (size_t)BATCH * HID * sizeof(float), hipMemcpyDeviceToDevice, stream);
  (void)hipMemcpyAsync(cT, cbuf,
                       (size_t)BATCH * HID * sizeof(float), hipMemcpyDeviceToDevice, stream);
}
